// GCN_24970939859460
// MI455X (gfx1250) — matrix operand order repaired and checked by reading, not compiled, not run on hardware
//
#include <hip/hip_runtime.h>
#include <hip/hip_bf16.h>
#include <stdint.h>

#define N_ATOMS_C 100000
#define N_BONDS_C 200000
#define MAX_NB_C  8
#define ATOM_F_C  133
#define BOND_IN_C 147      // ATOM_F + BOND_F
#define HID_C     256
#define FFN_C     1024
#define N_MOLS_C  4000

typedef __attribute__((ext_vector_type(16))) __bf16       v16bf;
typedef __attribute__((ext_vector_type(8)))  float        v8f;
typedef __attribute__((ext_vector_type(8)))  unsigned int v8u;

// round-to-nearest-even fp32 -> bf16 pair packed into one dword (low = even col)
__device__ __forceinline__ uint32_t pack2bf(float a, float b) {
  uint32_t ua = __float_as_uint(a), ub = __float_as_uint(b);
  ua = (ua + 0x7FFFu + ((ua >> 16) & 1u)) >> 16;
  ub = (ub + 0x7FFFu + ((ub >> 16) & 1u)) >> 16;
  return ua | (ub << 16);
}

__device__ __forceinline__ v8u mkv8(uint4 lo, uint4 hi) {
  v8u r;
  r[0] = lo.x; r[1] = lo.y; r[2] = lo.z; r[3] = lo.w;
  r[4] = hi.x; r[5] = hi.y; r[6] = hi.z; r[7] = hi.w;
  return r;
}

// One 16x16 f32 tile: A-strip (16 rows, K=32*S bf16) from LDS, B from pre-swizzled
// global weight image; B fragments are software double-buffered so each WMMA
// overlaps the next step's global_load_b128 pair. lane = h*16+ln.
template <int S, int LDA>
__device__ __forceinline__ v8f mm_tile(const uint32_t* __restrict__ As,
                                       const uint32_t* __restrict__ Wsw,
                                       int ln, int h, int nt) {
  v8f c = {0.f, 0.f, 0.f, 0.f, 0.f, 0.f, 0.f, 0.f};
  const int lane = h * 16 + ln;
  const uint32_t* pb = Wsw + ((size_t)(nt * 32 + lane)) * 8;
  uint4 b0 = *(const uint4*)pb;
  uint4 b1 = *(const uint4*)(pb + 4);
#pragma unroll
  for (int s = 0; s < S; ++s) {
    // A frag (ISA 16-bit A 16x32 layout): dwords {0..3, 8..11} + 4h + 16s of row ln
    const uint32_t* pa = As + ln * LDA + s * 16 + h * 4;
    uint4 a0 = *(const uint4*)pa;
    uint4 a1 = *(const uint4*)(pa + 8);
    uint4 nb0 = b0, nb1 = b1;
    if (s + 1 < S) {  // prefetch next B fragment
      const uint32_t* pn = Wsw + ((size_t)(((s + 1) * 16 + nt) * 32 + lane)) * 8;
      nb0 = *(const uint4*)pn;
      nb1 = *(const uint4*)(pn + 4);
    }
    v16bf a = __builtin_bit_cast(v16bf, mkv8(a0, a1));
    v16bf b = __builtin_bit_cast(v16bf, mkv8(b0, b1));
    c = __builtin_amdgcn_wmma_f32_16x16x32_bf16(false, a, false, b, (short)0, c,
                                                false, false);
    b0 = nb0; b1 = nb1;
  }
  return c;
}

// Convert row-major fp32 W[Kreal x 256] to bf16 B-fragment image (K padded to 32*S).
// Image index = ((s*16 + nt)*32 + lane)*8 + v ; lane=(h*16+ln): n=nt*16+ln, K=32s+16(v>>2)+8h+2(v&3){+1}.
__global__ void k_convert_w(const float* __restrict__ W, uint32_t* __restrict__ Wsw,
                            int Kreal, int S) {
  int id = blockIdx.x * blockDim.x + threadIdx.x;
  int total = S * 16 * 32 * 8;
  if (id >= total) return;
  int v  = id & 7;
  int t  = (id >> 3) & 31;
  int nt = (id >> 8) & 15;
  int s  = id >> 12;
  int ln = t & 15, h = t >> 4;
  int n  = nt * 16 + ln;
  int k0 = s * 32 + (v >> 2) * 16 + h * 8 + 2 * (v & 3);
  float x = (k0     < Kreal) ? W[(size_t)k0       * HID_C + n] : 0.f;
  float y = (k0 + 1 < Kreal) ? W[(size_t)(k0 + 1) * HID_C + n] : 0.f;
  Wsw[id] = pack2bf(x, y);
}

// ---------------- init: nei_input = fbonds @ W_i ; msg = relu(nei_input) -------------
__global__ void k_gemm_init(const float* __restrict__ fbonds,
                            const uint32_t* __restrict__ Wsw,
                            float* __restrict__ nei_input,
                            float* __restrict__ msg_out) {
  extern __shared__ uint32_t sm[];
  const int LDA = 84, S = 5;                  // K padded 147 -> 160 (80 dwords)
  uint32_t* As = sm;
  int tid = threadIdx.x;
  size_t r0 = (size_t)blockIdx.x * 128;
  for (int i = tid; i < 128 * 80; i += 256) {
    int row = i / 80, d = i - row * 80;
    int c0 = 2 * d;
    size_t grow = r0 + row;
    float x = 0.f, y = 0.f;
    if (grow < (size_t)N_BONDS_C) {
      const float* fr = fbonds + grow * (size_t)BOND_IN_C;
      if (c0     < BOND_IN_C) x = fr[c0];
      if (c0 + 1 < BOND_IN_C) y = fr[c0 + 1];
    }
    As[row * LDA + d] = pack2bf(x, y);
  }
  __syncthreads();
  int wave = tid >> 5, lane = tid & 31, ln = lane & 15, h = lane >> 4;
  const uint32_t* Aw = As + wave * 16 * LDA;
  size_t rbase = r0 + wave * 16 + 8 * h;
  if (r0 + 128 <= (size_t)N_BONDS_C) {        // fast path: whole tile in bounds
    for (int nt = 0; nt < 16; ++nt) {
      v8f c = mm_tile<S, LDA>(Aw, Wsw, ln, h, nt);
      int n = nt * 16 + ln;
#pragma unroll
      for (int r = 0; r < 8; ++r) {
        size_t o = (rbase + r) * HID_C + n;
        float v = c[r];
        nei_input[o] = v;
        msg_out[o]   = fmaxf(v, 0.f);
      }
    }
  } else {
    for (int nt = 0; nt < 16; ++nt) {
      v8f c = mm_tile<S, LDA>(Aw, Wsw, ln, h, nt);
      int n = nt * 16 + ln;
#pragma unroll
      for (int r = 0; r < 8; ++r) {
        size_t grow = rbase + r;
        if (grow < (size_t)N_BONDS_C) {
          size_t o = grow * HID_C + n;
          float v = c[r];
          nei_input[o] = v;
          msg_out[o]   = fmaxf(v, 0.f);
        }
      }
    }
  }
}

// -------- layer: msg_out = relu(nei_input + (gather_sum(msg_in, bgraph)) @ W_h) ------
__global__ void k_gemm_layer(const float* __restrict__ msg_in,
                             const int* __restrict__ bgraph,
                             const uint32_t* __restrict__ Wsw,
                             const float* __restrict__ nei_input,
                             float* __restrict__ msg_out) {
  extern __shared__ uint32_t sm[];
  const int LDA = 132, S = 8;                 // K = 256 (128 dwords, padded stride)
  uint32_t* As = sm;
  int* sidx = (int*)(sm + 128 * LDA);
  int tid = threadIdx.x;
  size_t r0 = (size_t)blockIdx.x * 128;
  for (int i = tid; i < 128 * MAX_NB_C; i += 256) {
    int row = i >> 3, j = i & 7;
    size_t grow = r0 + row;
    sidx[i] = (grow < (size_t)N_BONDS_C) ? bgraph[grow * MAX_NB_C + j] : 0;
  }
  __syncthreads();
  for (int i = tid; i < 128 * 64; i += 256) {
    int row = i >> 6, cv = i & 63;
    const int* nb = sidx + row * MAX_NB_C;
    float ax = 0.f, ay = 0.f, az = 0.f, aw = 0.f;
#pragma unroll
    for (int j = 0; j < MAX_NB_C; ++j) {
      float4 vv = *((const float4*)(msg_in + (size_t)nb[j] * HID_C) + cv);
      ax += vv.x; ay += vv.y; az += vv.z; aw += vv.w;
    }
    As[row * LDA + 2 * cv]     = pack2bf(ax, ay);
    As[row * LDA + 2 * cv + 1] = pack2bf(az, aw);
  }
  __syncthreads();
  int wave = tid >> 5, lane = tid & 31, ln = lane & 15, h = lane >> 4;
  const uint32_t* Aw = As + wave * 16 * LDA;
  size_t rbase = r0 + wave * 16 + 8 * h;
  if (r0 + 128 <= (size_t)N_BONDS_C) {        // fast path
    for (int nt = 0; nt < 16; ++nt) {
      v8f c = mm_tile<S, LDA>(Aw, Wsw, ln, h, nt);
      int n = nt * 16 + ln;
#pragma unroll
      for (int r = 0; r < 8; ++r) {
        size_t o = (rbase + r) * HID_C + n;
        msg_out[o] = fmaxf(nei_input[o] + c[r], 0.f);
      }
    }
  } else {
    for (int nt = 0; nt < 16; ++nt) {
      v8f c = mm_tile<S, LDA>(Aw, Wsw, ln, h, nt);
      int n = nt * 16 + ln;
#pragma unroll
      for (int r = 0; r < 8; ++r) {
        size_t grow = rbase + r;
        if (grow < (size_t)N_BONDS_C) {
          size_t o = grow * HID_C + n;
          msg_out[o] = fmaxf(nei_input[o] + c[r], 0.f);
        }
      }
    }
  }
}

// -- readout: atom_h = concat(fatoms, gather_sum(msg, agraph)) @ W_o + b_o -> d_out ---
__global__ void k_readout(const float* __restrict__ fatoms,
                          const int* __restrict__ agraph,
                          const float* __restrict__ msg,
                          const uint32_t* __restrict__ Wsw,
                          const float* __restrict__ b_o,
                          float* __restrict__ out) {
  extern __shared__ uint32_t sm[];
  const int LDA = 212, S = 13;                // K = 389 padded -> 416 (208 dwords)
  uint32_t* As = sm;
  int* sidx = (int*)(sm + 128 * LDA);
  int tid = threadIdx.x;
  size_t r0 = (size_t)blockIdx.x * 128;
  for (int i = tid; i < 128 * MAX_NB_C; i += 256) {
    int row = i >> 3, j = i & 7;
    size_t grow = r0 + row;
    sidx[i] = (grow < (size_t)N_ATOMS_C) ? agraph[grow * MAX_NB_C + j] : 0;
  }
  __syncthreads();
  for (int i = tid; i < 128 * 208; i += 256) {
    int row = i / 208, d = i - row * 208;
    size_t grow = r0 + row;
    const int* nb = sidx + row * MAX_NB_C;
    float vals[2];
#pragma unroll
    for (int e = 0; e < 2; ++e) {
      int col = 2 * d + e;
      float v = 0.f;
      if (grow < (size_t)N_ATOMS_C) {
        if (col < ATOM_F_C) {
          v = fatoms[grow * (size_t)ATOM_F_C + col];
        } else if (col < ATOM_F_C + HID_C) {
          int k = col - ATOM_F_C;
          float s = 0.f;
#pragma unroll
          for (int j = 0; j < MAX_NB_C; ++j)
            s += msg[(size_t)nb[j] * HID_C + k];
          v = s;
        }
      }
      vals[e] = v;
    }
    As[row * LDA + d] = pack2bf(vals[0], vals[1]);
  }
  __syncthreads();
  int wave = tid >> 5, lane = tid & 31, ln = lane & 15, h = lane >> 4;
  const uint32_t* Aw = As + wave * 16 * LDA;
  size_t rbase = r0 + wave * 16 + 8 * h;
  if (r0 + 128 <= (size_t)N_ATOMS_C) {        // fast path
    for (int nt = 0; nt < 16; ++nt) {
      v8f c = mm_tile<S, LDA>(Aw, Wsw, ln, h, nt);
      int n = nt * 16 + ln;
      float bn = b_o[n];
#pragma unroll
      for (int r = 0; r < 8; ++r)
        out[(rbase + r) * HID_C + n] = c[r] + bn;
    }
  } else {
    for (int nt = 0; nt < 16; ++nt) {
      v8f c = mm_tile<S, LDA>(Aw, Wsw, ln, h, nt);
      int n = nt * 16 + ln;
      float bn = b_o[n];
#pragma unroll
      for (int r = 0; r < 8; ++r) {
        size_t grow = rbase + r;
        if (grow < (size_t)N_ATOMS_C)
          out[grow * HID_C + n] = c[r] + bn;
      }
    }
  }
}

// ------------- per-molecule mean over sorted mol_ids (binary search) -----------------
__global__ void k_seg_mean(const float* __restrict__ atom_h,
                           const int* __restrict__ mol_ids,
                           float* __restrict__ mol_mean) {
  int m = blockIdx.x, c = threadIdx.x;
  int lo = 0, hi = N_ATOMS_C;
  while (lo < hi) { int mid = (lo + hi) >> 1; if (mol_ids[mid] < m) lo = mid + 1; else hi = mid; }
  int start = lo;
  hi = N_ATOMS_C;
  while (lo < hi) { int mid = (lo + hi) >> 1; if (mol_ids[mid] < m + 1) lo = mid + 1; else hi = mid; }
  int end = lo;
  float s = 0.f;
  for (int a = start; a < end; ++a) s += atom_h[(size_t)a * HID_C + c];
  float cnt = (float)(end - start);
  mol_mean[(size_t)m * HID_C + c] = s / fmaxf(cnt, 1.f);
}

// ---------- mol FFN: relu(mean @ W_mol_h + b) @ W_mol_o + b -> out[m] ----------------
__global__ void k_mol_ffn(const float* __restrict__ mol_mean,
                          const float* __restrict__ Wh, const float* __restrict__ bh,
                          const float* __restrict__ Wo, const float* __restrict__ bo,
                          float* __restrict__ out) {
  __shared__ float x[HID_C];
  __shared__ float red[256];
  int m = blockIdx.x, t = threadIdx.x;
  x[t] = mol_mean[(size_t)m * HID_C + t];
  __syncthreads();
  float part = 0.f;
#pragma unroll
  for (int q = 0; q < 4; ++q) {
    int col = q * 256 + t;
    float hv = bh[col];
    for (int k = 0; k < HID_C; ++k) hv = fmaf(x[k], Wh[(size_t)k * FFN_C + col], hv);
    hv = fmaxf(hv, 0.f);
    part = fmaf(hv, Wo[col], part);
  }
  red[t] = part;
  __syncthreads();
  for (int off = 128; off > 0; off >>= 1) {
    if (t < off) red[t] += red[t + off];
    __syncthreads();
  }
  if (t == 0) out[m] = red[0] + bo[0];
}

extern "C" void kernel_launch(void* const* d_in, const int* in_sizes, int n_in,
                              void* d_out, int out_size, void* d_ws, size_t ws_size,
                              hipStream_t stream) {
  (void)in_sizes; (void)n_in; (void)out_size; (void)ws_size;
  const float* fatoms  = (const float*)d_in[0];
  const float* fbonds  = (const float*)d_in[1];
  const int*   agraph  = (const int*)d_in[2];
  const int*   bgraph  = (const int*)d_in[3];
  const int*   mol_ids = (const int*)d_in[4];
  const float* W_i     = (const float*)d_in[6];
  const float* W_h     = (const float*)d_in[7];
  const float* W_o     = (const float*)d_in[8];
  const float* b_o     = (const float*)d_in[9];
  const float* W_mol_h = (const float*)d_in[10];
  const float* b_mol_h = (const float*)d_in[11];
  const float* W_mol_o = (const float*)d_in[12];
  const float* b_mol_o = (const float*)d_in[13];
  float* out = (float*)d_out;

  const size_t MSG = (size_t)N_BONDS_C * HID_C;
  float* nei_input = (float*)d_ws;
  float* msgA = nei_input + MSG;
  float* msgB = msgA + MSG;
  uint32_t* WiSw = (uint32_t*)(msgB + MSG);
  uint32_t* WhSw = WiSw + 5 * 4096;
  uint32_t* WoSw = WhSw + 8 * 4096;
  float* mol_mean = (float*)(WoSw + 13 * 4096);

  // weight pre-swizzle to bf16 fragment layout
  k_convert_w<<<(5 * 4096 + 255) / 256, 256, 0, stream>>>(W_i, WiSw, BOND_IN_C, 5);
  k_convert_w<<<(8 * 4096 + 255) / 256, 256, 0, stream>>>(W_h, WhSw, HID_C, 8);
  k_convert_w<<<(13 * 4096 + 255) / 256, 256, 0, stream>>>(W_o, WoSw, ATOM_F_C + HID_C, 13);

  const int BBLK = (N_BONDS_C + 127) / 128;   // 1563
  const int ABLK = (N_ATOMS_C + 127) / 128;   // 782
  const size_t SH_INIT  = 128 * 84  * sizeof(uint32_t);
  const size_t SH_LAYER = 128 * 132 * sizeof(uint32_t) + 128 * MAX_NB_C * sizeof(int);
  const size_t SH_READ  = 128 * 212 * sizeof(uint32_t) + 128 * MAX_NB_C * sizeof(int);

  k_gemm_init<<<BBLK, 256, SH_INIT, stream>>>(fbonds, WiSw, nei_input, msgA);
  // 4 message-passing layers, ping-pong A<->B (ends in msgA)
  k_gemm_layer<<<BBLK, 256, SH_LAYER, stream>>>(msgA, bgraph, WhSw, nei_input, msgB);
  k_gemm_layer<<<BBLK, 256, SH_LAYER, stream>>>(msgB, bgraph, WhSw, nei_input, msgA);
  k_gemm_layer<<<BBLK, 256, SH_LAYER, stream>>>(msgA, bgraph, WhSw, nei_input, msgB);
  k_gemm_layer<<<BBLK, 256, SH_LAYER, stream>>>(msgB, bgraph, WhSw, nei_input, msgA);

  k_readout<<<ABLK, 256, SH_READ, stream>>>(fatoms, agraph, msgA, WoSw, b_o, out);
  k_seg_mean<<<N_MOLS_C, 256, 0, stream>>>(out, mol_ids, mol_mean);
  k_mol_ffn<<<N_MOLS_C, 256, 0, stream>>>(mol_mean, W_mol_h, b_mol_h, W_mol_o,
                                          b_mol_o, out + (size_t)N_ATOMS_C * HID_C);
}
